// CRF_81939386073425
// MI455X (gfx1250) — compile-verified, matrix operand order checked by reading, not run on hardware
//
#include <hip/hip_runtime.h>
#include <hip/hip_bf16.h>

typedef __attribute__((ext_vector_type(16))) _Float16 v16h;
typedef __attribute__((ext_vector_type(8)))  float    v8f;
typedef __attribute__((ext_vector_type(4)))  float    v4f;

#define BATCH 64
#define TLEN  512
#define EDIM  2048
#define KTAG  32
#define NROWS (BATCH * TLEN)          // 32768 rows of the GEMM
#define ECHUNKS (EDIM / 32)           // 64 k-chunks of 32
#define LOG2E 1.4426950408889634f
#define LN2   0.6931471805599453f

// ---------------------------------------------------------------------------
// Kernel 0: pack W (E x K, fp32, row-major) into f16 WMMA B-fragments.
// Fragment layout per k-chunk c and n-tile nt (cols nt*16..nt*16+15):
//   lane l holds column n = nt*16 + (l&15); kb = (l>>4)*8
//   halves 0..7  -> k = c*32 + kb + h
//   halves 8..15 -> k = c*32 + kb + 16 + (h-8)
// Stored contiguously: 16 halves (32B) per lane -> single 32B load later.
// ---------------------------------------------------------------------------
__global__ void crf_pack_w(const float* __restrict__ W,
                           _Float16* __restrict__ bfrag) {
    int tid = blockIdx.x * blockDim.x + threadIdx.x;      // 0..4095
    if (tid >= ECHUNKS * 2 * 32) return;
    int lane = tid & 31;
    int nt   = (tid >> 5) & 1;
    int c    = tid >> 6;
    int n  = nt * 16 + (lane & 15);
    int kb = (lane >> 4) * 8;
    _Float16* dst = bfrag + (size_t)((c * 2 + nt) * 32 + lane) * 16;
#pragma unroll
    for (int h = 0; h < 16; ++h) {
        int e = c * 32 + kb + (h < 8 ? h : h + 8);
        dst[h] = (_Float16)W[e * KTAG + n];
    }
}

// ---------------------------------------------------------------------------
// Kernel 1: logits = X @ W + b via v_wmma_f32_16x16x32_f16.
// 256 threads = 8 waves; each wave owns a 16(M) x 32(N) tile; block = 128 rows.
// X is streamed exactly once -> non-temporal loads (keep L2 for W frags/logits).
// ---------------------------------------------------------------------------
__global__ void __launch_bounds__(256)
crf_gemm(const float* __restrict__ X,
         const _Float16* __restrict__ bfrag,
         const float* __restrict__ bias,
         float* __restrict__ logits) {
    const int lane = threadIdx.x & 31;
    const int wave = threadIdx.x >> 5;
    const int rowbase = blockIdx.x * 128 + wave * 16;

    const int rl = lane & 15;           // local row (A) / local col (D)
    const int hi = lane >> 4;           // half-wave select
    const int kb = hi * 8;              // k sub-offset for A/B fragments

    const float* __restrict__ pX = X + (size_t)(rowbase + rl) * EDIM;

    v8f acc0 = {};
    v8f acc1 = {};

#pragma unroll 2
    for (int c = 0; c < ECHUNKS; ++c) {
        const int e0 = c * 32 + kb;
        // one full 128B cacheline per row per chunk across the wave
        const v4f f0 = __builtin_nontemporal_load((const v4f*)(pX + e0));
        const v4f f1 = __builtin_nontemporal_load((const v4f*)(pX + e0 + 4));
        const v4f f2 = __builtin_nontemporal_load((const v4f*)(pX + e0 + 16));
        const v4f f3 = __builtin_nontemporal_load((const v4f*)(pX + e0 + 20));
        // unconditional speculative prefetch 4 chunks ahead (OOB is dropped)
        __builtin_prefetch(pX + e0 + 128, 0, 1);   // global_prefetch_b8

        v16h a;
        a[0]  = (_Float16)f0[0];  a[1]  = (_Float16)f0[1];
        a[2]  = (_Float16)f0[2];  a[3]  = (_Float16)f0[3];
        a[4]  = (_Float16)f1[0];  a[5]  = (_Float16)f1[1];
        a[6]  = (_Float16)f1[2];  a[7]  = (_Float16)f1[3];
        a[8]  = (_Float16)f2[0];  a[9]  = (_Float16)f2[1];
        a[10] = (_Float16)f2[2];  a[11] = (_Float16)f2[3];
        a[12] = (_Float16)f3[0];  a[13] = (_Float16)f3[1];
        a[14] = (_Float16)f3[2];  a[15] = (_Float16)f3[3];

        const v16h b0 = *(const v16h*)(bfrag + (size_t)((c * 2 + 0) * 32 + lane) * 16);
        const v16h b1 = *(const v16h*)(bfrag + (size_t)((c * 2 + 1) * 32 + lane) * 16);

        acc0 = __builtin_amdgcn_wmma_f32_16x16x32_f16(false, a, false, b0,
                                                      (short)0, acc0, false, false);
        acc1 = __builtin_amdgcn_wmma_f32_16x16x32_f16(false, a, false, b1,
                                                      (short)0, acc1, false, false);
    }

    // D layout: VGPR r -> (M = hi*8 + r, N = lane&15) within each 16-col tile
#pragma unroll
    for (int r = 0; r < 8; ++r) {
        const int row = rowbase + hi * 8 + r;
        logits[(size_t)row * KTAG + rl]      = acc0[r] + bias[rl];
        logits[(size_t)row * KTAG + 16 + rl] = acc1[r] + bias[16 + rl];
    }
}

// ---------------------------------------------------------------------------
// Kernel 2: joint (numerator) score, one wave per batch.
// ---------------------------------------------------------------------------
__global__ void __launch_bounds__(32)
crf_joint(const float* __restrict__ logits,
          const int* __restrict__ y,
          const unsigned char* __restrict__ mask,
          const float* __restrict__ trans,
          const float* __restrict__ start,
          const float* __restrict__ endt,
          float* __restrict__ log_num) {
    const int b = blockIdx.x;
    const int lane = threadIdx.x;
    const int* yb = y + b * TLEN;
    const unsigned char* mb = mask + b * TLEN;
    const float* lb = logits + (size_t)b * TLEN * KTAG;

    float emit = 0.f, tsc = 0.f;
    int len = 0;
    for (int t = lane; t < TLEN; t += 32) {
        if (mb[t]) {
            ++len;
            emit += lb[(size_t)t * KTAG + yb[t]];
            if (t >= 1) tsc += trans[yb[t - 1] * KTAG + yb[t]];
        }
    }
    float s = emit + tsc;
#pragma unroll
    for (int off = 16; off > 0; off >>= 1) {
        s   += __shfl_xor(s, off, 32);
        len += __shfl_xor(len, off, 32);
    }
    if (lane == 0)
        log_num[b] = start[yb[0]] + s + endt[yb[len - 1]];
}

// ---------------------------------------------------------------------------
// Kernel 3: forward algorithm (denominator), one wave per batch, lane = state.
// ---------------------------------------------------------------------------
__global__ void __launch_bounds__(32)
crf_forward(const float* __restrict__ logits,
            const unsigned char* __restrict__ mask,
            const float* __restrict__ trans,
            const float* __restrict__ start,
            const float* __restrict__ endt,
            float* __restrict__ log_den) {
    const int b = blockIdx.x;
    const int lane = threadIdx.x;
    const unsigned char* mb = mask + b * TLEN;
    const float* lb = logits + (size_t)b * TLEN * KTAG;

    // transitions column for this destination state (32 VGPRs)
    float tcol[KTAG];
#pragma unroll
    for (int k = 0; k < KTAG; ++k) tcol[k] = trans[k * KTAG + lane];

    // sequence length (mask is a contiguous prefix)
    int len = 0;
    for (int t = lane; t < TLEN; t += 32) len += mb[t] ? 1 : 0;
#pragma unroll
    for (int off = 16; off > 0; off >>= 1) len += __shfl_xor(len, off, 32);

    float alpha = start[lane] + lb[lane];

    for (int t = 1; t < len; ++t) {
        const float x = lb[(size_t)t * KTAG + lane];
        float v[KTAG];
        float m = -__builtin_inff();
#pragma unroll
        for (int k = 0; k < KTAG; ++k) {
            const float ak = __int_as_float(
                __builtin_amdgcn_readlane(__float_as_int(alpha), k));
            v[k] = ak + tcol[k];
            m = fmaxf(m, v[k]);
        }
        float s = 0.f;
#pragma unroll
        for (int k = 0; k < KTAG; ++k)
            s += __builtin_amdgcn_exp2f((v[k] - m) * LOG2E);
        alpha = m + __builtin_amdgcn_logf(s) * LN2 + x;  // logf builtin = log2
    }

    // wave-wide logsumexp(alpha + end)
    float val = alpha + endt[lane];
    float m = val;
#pragma unroll
    for (int off = 16; off > 0; off >>= 1) m = fmaxf(m, __shfl_xor(m, off, 32));
    float s = __builtin_amdgcn_exp2f((val - m) * LOG2E);
#pragma unroll
    for (int off = 16; off > 0; off >>= 1) s += __shfl_xor(s, off, 32);
    if (lane == 0) log_den[b] = m + __builtin_amdgcn_logf(s) * LN2;
}

// ---------------------------------------------------------------------------
// Kernel 4: final scalar reduction: out = -sum_b(log_num - log_den)
// ---------------------------------------------------------------------------
__global__ void __launch_bounds__(32)
crf_reduce(const float* __restrict__ log_num,
           const float* __restrict__ log_den,
           float* __restrict__ out) {
    const int lane = threadIdx.x;
    float s = 0.f;
    for (int b = lane; b < BATCH; b += 32) s += log_num[b] - log_den[b];
#pragma unroll
    for (int off = 16; off > 0; off >>= 1) s += __shfl_xor(s, off, 32);
    if (lane == 0) out[0] = -s;
}

// ---------------------------------------------------------------------------
extern "C" void kernel_launch(void* const* d_in, const int* in_sizes, int n_in,
                              void* d_out, int out_size, void* d_ws, size_t ws_size,
                              hipStream_t stream) {
    const float*         X     = (const float*)d_in[0];
    const int*           y     = (const int*)d_in[1];
    const unsigned char* mask  = (const unsigned char*)d_in[2];
    const float*         W     = (const float*)d_in[3];
    const float*         bias  = (const float*)d_in[4];
    const float*         trans = (const float*)d_in[5];
    const float*         start = (const float*)d_in[6];
    const float*         endt  = (const float*)d_in[7];
    float*               out   = (float*)d_out;

    char* ws = (char*)d_ws;
    float*    logits  = (float*)ws;                                  // 4 MB
    _Float16* bfrag   = (_Float16*)(ws + (size_t)NROWS * KTAG * 4);  // 128 KB
    float*    log_num = (float*)(ws + (size_t)NROWS * KTAG * 4 + ECHUNKS * 2 * 32 * 16 * 2);
    float*    log_den = log_num + BATCH;

    crf_pack_w <<<16, 256, 0, stream>>>(W, bfrag);
    crf_gemm   <<<NROWS / 128, 256, 0, stream>>>(X, bfrag, bias, logits);
    crf_joint  <<<BATCH, 32, 0, stream>>>(logits, y, mask, trans, start, endt, log_num);
    crf_forward<<<BATCH, 32, 0, stream>>>(logits, mask, trans, start, endt, log_den);
    crf_reduce <<<1, 32, 0, stream>>>(log_num, log_den, out);
}
